// GARCHGRUCell_83605833383957
// MI455X (gfx1250) — matrix operand order repaired and checked by reading, not compiled, not run on hardware
//
#include <hip/hip_runtime.h>
#include <hip/hip_bf16.h>
#include <math.h>

#define BB   8192
#define INW  64
#define HH   1024
#define KTOT 1088          // IN + H, K of the fused [x|h] GEMMs
#define BM   128           // output rows per workgroup
#define BN   64            // output cols per workgroup
#define ASTRIDE 36         // LDS row stride (bf16 elems), padded vs 32
#define BSTRIDE 36

typedef __attribute__((ext_vector_type(16))) __bf16 v16bf;
typedef __attribute__((ext_vector_type(8)))  float  v8f;

// ---------- fp32 -> bf16 hi/lo split (round-to-nearest-even) ----------
__device__ __forceinline__ void split_bf(float f, unsigned short& hi, unsigned short& lo) {
    unsigned int u = __builtin_bit_cast(unsigned int, f);
    unsigned int r = u + 0x7FFFu + ((u >> 16) & 1u);
    hi = (unsigned short)(r >> 16);
    float fh = __builtin_bit_cast(float, (unsigned int)hi << 16);
    float res = f - fh;
    unsigned int u2 = __builtin_bit_cast(unsigned int, res);
    unsigned int r2 = u2 + 0x7FFFu + ((u2 >> 16) & 1u);
    lo = (unsigned short)(r2 >> 16);
}

__device__ __forceinline__ float sigm(float v) { return 1.f / (1.f + expf(-v)); }
__device__ __forceinline__ float softplusf(float v) { return (v > 20.f) ? v : log1pf(expf(v)); }

// ---------- WMMA wrapper ----------
__device__ __forceinline__ v8f wmma_bf16(v16bf a, v16bf b, v8f c) {
    return __builtin_amdgcn_wmma_f32_16x16x32_bf16(false, a, false, b, (short)0, c, false, false);
}

// ---------- CDNA5 async global->LDS copy (ASYNCcnt path, ISA 15.18.3 op 97) ----------
// per-lane: LDS[vdst] = MEM[vaddr..vaddr+7]
__device__ __forceinline__ void async_ld_b64(unsigned ldsOff, unsigned long long gAddr) {
    asm volatile("global_load_async_to_lds_b64 %0, %1, off"
                 :: "v"(ldsOff), "v"(gAddr) : "memory");
}
__device__ __forceinline__ void wait_async0() {
    asm volatile("s_wait_asynccnt 0x0" ::: "memory");
}

// ---------- fragment loaders from LDS (wave32 layouts per CDNA5 ISA 7.12.2) ----------
// A 16x32 bf16: lanes0-15 row M=lane hold K {0..7,16..23}; lanes16-31 hold K {8..15,24..31}
__device__ __forceinline__ v16bf load_a_frag(const unsigned short* s, int row0, int lane) {
    const int half = lane >> 4;
    const int m    = lane & 15;
    const unsigned short* p = s + (row0 + m) * ASTRIDE + half * 8;
    union { v16bf v; unsigned long long q[4]; } u;
    u.q[0] = *reinterpret_cast<const unsigned long long*>(p);
    u.q[1] = *reinterpret_cast<const unsigned long long*>(p + 4);
    u.q[2] = *reinterpret_cast<const unsigned long long*>(p + 16);
    u.q[3] = *reinterpret_cast<const unsigned long long*>(p + 20);
    return u.v;
}
// B 32x16 bf16: lane n=lane%16 is column; lanes0-15 hold K {0..7,16..23}, lanes16-31 K {8..15,24..31}.
// LDS tile is stored as [n][k] (weight W[n][k]) so a lane reads two runs of 8 contiguous bf16.
__device__ __forceinline__ v16bf load_b_frag(const unsigned short* s, int col0, int lane) {
    const int half = lane >> 4;
    const int n    = lane & 15;
    const unsigned short* p = s + (col0 + n) * BSTRIDE + half * 8;
    union { v16bf v; unsigned long long q[4]; } u;
    u.q[0] = *reinterpret_cast<const unsigned long long*>(p);
    u.q[1] = *reinterpret_cast<const unsigned long long*>(p + 4);
    u.q[2] = *reinterpret_cast<const unsigned long long*>(p + 16);
    u.q[3] = *reinterpret_cast<const unsigned long long*>(p + 20);
    return u.v;
}

// ---------- global(hi/lo planes, width KTOT) -> LDS stage of a rows x 32 chunk ----------
// Uses the CDNA5 async tensor/LDS DMA path: no VGPR round trip.
__device__ __forceinline__ void stage32(const unsigned short* __restrict__ ghi,
                                        const unsigned short* __restrict__ glo,
                                        unsigned short* shi, unsigned short* slo,
                                        long long gRowBase, int kBase, int rows, int tid) {
    const unsigned shOff = (unsigned)(uintptr_t)shi;   // generic->LDS offset (addr[31:0])
    const unsigned slOff = (unsigned)(uintptr_t)slo;
    const int slots = rows * 8;                        // 8 x u64 (4 bf16) segments per row
    for (int s = tid; s < slots; s += 256) {
        const int row = s >> 3, seg = s & 7;
        const long long goff = (gRowBase + row) * (long long)KTOT + kBase + seg * 4;
        const unsigned soff = (unsigned)(row * ASTRIDE + seg * 4) * 2u;  // bytes
        async_ld_b64(shOff + soff, (unsigned long long)(uintptr_t)(ghi + goff));
        async_ld_b64(slOff + soff, (unsigned long long)(uintptr_t)(glo + goff));
    }
}

// =====================================================================
// Kernel 0: pack [x|h_prev] and stacked [Wg|Ug] weights into bf16 hi/lo planes
// =====================================================================
__global__ void __launch_bounds__(256)
pack_kernel(const float* __restrict__ x, const float* __restrict__ h_prev,
            const float* __restrict__ Wz_w, const float* __restrict__ Uz_w,
            const float* __restrict__ Wr_w, const float* __restrict__ Ur_w,
            const float* __restrict__ Wh_w, const float* __restrict__ Uh_w,
            unsigned short* __restrict__ A1hi, unsigned short* __restrict__ A1lo,
            unsigned short* __restrict__ A3hi, unsigned short* __restrict__ A3lo,
            unsigned short* __restrict__ Whi,  unsigned short* __restrict__ Wlo) {
    const long long N1 = (long long)BB * KTOT;
    const long long N2 = 3LL * HH * KTOT;
    for (long long idx = (long long)blockIdx.x * blockDim.x + threadIdx.x;
         idx < N1 + N2; idx += (long long)gridDim.x * blockDim.x) {
        unsigned short hi, lo;
        if (idx < N1) {
            const int i = (int)(idx / KTOT), k = (int)(idx % KTOT);
            const float f = (k < INW) ? x[(long long)i * INW + k]
                                      : h_prev[(long long)i * HH + (k - INW)];
            split_bf(f, hi, lo);
            A1hi[idx] = hi; A1lo[idx] = lo;
            if (k < INW) { A3hi[idx] = hi; A3lo[idx] = lo; }   // r*h cols filled later
        } else {
            const long long j = idx - N1;
            const int g = (int)(j / ((long long)HH * KTOT));
            const long long rem = j % ((long long)HH * KTOT);
            const int n = (int)(rem / KTOT), k = (int)(rem % KTOT);
            const float* Wp = (g == 0) ? Wz_w : (g == 1) ? Wr_w : Wh_w;
            const float* Up = (g == 0) ? Uz_w : (g == 1) ? Ur_w : Uh_w;
            const float f = (k < INW) ? Wp[(long long)n * INW + k]
                                      : Up[(long long)n * HH + (k - INW)];
            split_bf(f, hi, lo);
            Whi[j] = hi; Wlo[j] = lo;
        }
    }
}

// =====================================================================
// Kernel 1: z-pre & r-pre GEMMs (shared A), epilogue -> z, r*h_prev (split)
// =====================================================================
__global__ void __launch_bounds__(256)
gates_kernel(const unsigned short* __restrict__ A1hi, const unsigned short* __restrict__ A1lo,
             const unsigned short* __restrict__ Whi,  const unsigned short* __restrict__ Wlo,
             const float* __restrict__ Wz_b, const float* __restrict__ Uz_b,
             const float* __restrict__ Wr_b, const float* __restrict__ Ur_b,
             const float* __restrict__ h_prev,
             float* __restrict__ zbuf,
             unsigned short* __restrict__ A3hi, unsigned short* __restrict__ A3lo) {
    __shared__ unsigned short Ash[BM * ASTRIDE], Asl[BM * ASTRIDE];
    __shared__ unsigned short Bzh[BN * BSTRIDE], Bzl[BN * BSTRIDE];
    __shared__ unsigned short Brh[BN * BSTRIDE], Brl[BN * BSTRIDE];

    const int tid  = threadIdx.x;
    const int lane = tid & 31, wave = tid >> 5;
    const int wr = wave >> 1, wc = wave & 1;          // 4x2 waves -> 128x64 tile
    const long long rowBlock = (long long)blockIdx.y * BM;
    const int colBlock = blockIdx.x * BN;

    const unsigned short* WzH = Whi;
    const unsigned short* WzL = Wlo;
    const unsigned short* WrH = Whi + (long long)HH * KTOT;
    const unsigned short* WrL = Wlo + (long long)HH * KTOT;

    v8f accZ[2][2], accR[2][2];
    const v8f vzero = {0.f, 0.f, 0.f, 0.f, 0.f, 0.f, 0.f, 0.f};
#pragma unroll
    for (int i = 0; i < 2; ++i)
#pragma unroll
        for (int j = 0; j < 2; ++j) { accZ[i][j] = vzero; accR[i][j] = vzero; }

    for (int kc = 0; kc < KTOT; kc += 32) {
        stage32(A1hi, A1lo, Ash, Asl, rowBlock, kc, BM, tid);
        stage32(WzH,  WzL,  Bzh, Bzl, colBlock, kc, BN, tid);
        stage32(WrH,  WrL,  Brh, Brl, colBlock, kc, BN, tid);
        wait_async0();
        __syncthreads();

        v16bf ah[2], al[2];
#pragma unroll
        for (int t = 0; t < 2; ++t) {
            ah[t] = load_a_frag(Ash, wr * 32 + t * 16, lane);
            al[t] = load_a_frag(Asl, wr * 32 + t * 16, lane);
        }
#pragma unroll
        for (int t = 0; t < 2; ++t) {
            v16bf bh = load_b_frag(Bzh, wc * 32 + t * 16, lane);
            v16bf bl = load_b_frag(Bzl, wc * 32 + t * 16, lane);
#pragma unroll
            for (int i = 0; i < 2; ++i) {
                accZ[i][t] = wmma_bf16(ah[i], bh, accZ[i][t]);
                accZ[i][t] = wmma_bf16(ah[i], bl, accZ[i][t]);
                accZ[i][t] = wmma_bf16(al[i], bh, accZ[i][t]);
            }
            bh = load_b_frag(Brh, wc * 32 + t * 16, lane);
            bl = load_b_frag(Brl, wc * 32 + t * 16, lane);
#pragma unroll
            for (int i = 0; i < 2; ++i) {
                accR[i][t] = wmma_bf16(ah[i], bh, accR[i][t]);
                accR[i][t] = wmma_bf16(ah[i], bl, accR[i][t]);
                accR[i][t] = wmma_bf16(al[i], bh, accR[i][t]);
            }
        }
        __syncthreads();
    }

    // epilogue: z = sigmoid(zpre + biases), r*h_prev -> bf16 hi/lo into A3 planes
    const int half = lane >> 4, n = lane & 15;
#pragma unroll
    for (int ti = 0; ti < 2; ++ti)
#pragma unroll
        for (int tj = 0; tj < 2; ++tj) {
            const int gcol = colBlock + wc * 32 + tj * 16 + n;
            const float zb = Wz_b[gcol] + Uz_b[gcol];
            const float rb = Wr_b[gcol] + Ur_b[gcol];
#pragma unroll
            for (int r = 0; r < 8; ++r) {
                const long long grow = rowBlock + wr * 32 + ti * 16 + half * 8 + r;
                const float z  = sigm(accZ[ti][tj][r] + zb);
                const float rr = sigm(accR[ti][tj][r] + rb);
                const float hp = h_prev[grow * HH + gcol];
                const float rh = rr * hp;
                zbuf[grow * HH + gcol] = z;
                unsigned short hi, lo;
                split_bf(rh, hi, lo);
                const long long aidx = grow * KTOT + INW + gcol;
                A3hi[aidx] = hi; A3lo[aidx] = lo;
            }
        }
}

// =====================================================================
// Kernel 2: h-tilde GEMM [x|r*h] @ [Wh|Uh].T + fully fused GRU/GARCH epilogue -> h
// =====================================================================
__global__ void __launch_bounds__(256)
update_kernel(const unsigned short* __restrict__ A3hi, const unsigned short* __restrict__ A3lo,
              const unsigned short* __restrict__ Whi,  const unsigned short* __restrict__ Wlo,
              const float* __restrict__ Wh_b, const float* __restrict__ Uh_b,
              const float* __restrict__ Wg_w, const float* __restrict__ Wg_b,
              const float* __restrict__ h_prev, const float* __restrict__ zbuf,
              const float* __restrict__ eps_prev, const float* __restrict__ sig_prev,
              const float* __restrict__ omega_raw, const float* __restrict__ alpha_raw,
              const float* __restrict__ beta_raw, const float* __restrict__ gamma_p,
              float* __restrict__ out_h) {
    __shared__ unsigned short Ash[BM * ASTRIDE], Asl[BM * ASTRIDE];
    __shared__ unsigned short Bhh[BN * BSTRIDE], Bhl[BN * BSTRIDE];

    const int tid  = threadIdx.x;
    const int lane = tid & 31, wave = tid >> 5;
    const int wr = wave >> 1, wc = wave & 1;
    const long long rowBlock = (long long)blockIdx.y * BM;
    const int colBlock = blockIdx.x * BN;

    const unsigned short* WhH = Whi + 2LL * HH * KTOT;
    const unsigned short* WhL = Wlo + 2LL * HH * KTOT;

    v8f acc[2][2];
    const v8f vzero = {0.f, 0.f, 0.f, 0.f, 0.f, 0.f, 0.f, 0.f};
#pragma unroll
    for (int i = 0; i < 2; ++i)
#pragma unroll
        for (int j = 0; j < 2; ++j) acc[i][j] = vzero;

    for (int kc = 0; kc < KTOT; kc += 32) {
        stage32(A3hi, A3lo, Ash, Asl, rowBlock, kc, BM, tid);
        stage32(WhH,  WhL,  Bhh, Bhl, colBlock, kc, BN, tid);
        wait_async0();
        __syncthreads();

        v16bf ah[2], al[2];
#pragma unroll
        for (int t = 0; t < 2; ++t) {
            ah[t] = load_a_frag(Ash, wr * 32 + t * 16, lane);
            al[t] = load_a_frag(Asl, wr * 32 + t * 16, lane);
        }
#pragma unroll
        for (int t = 0; t < 2; ++t) {
            const v16bf bh = load_b_frag(Bhh, wc * 32 + t * 16, lane);
            const v16bf bl = load_b_frag(Bhl, wc * 32 + t * 16, lane);
#pragma unroll
            for (int i = 0; i < 2; ++i) {
                acc[i][t] = wmma_bf16(ah[i], bh, acc[i][t]);
                acc[i][t] = wmma_bf16(ah[i], bl, acc[i][t]);
                acc[i][t] = wmma_bf16(al[i], bh, acc[i][t]);
            }
        }
        __syncthreads();
    }

    // GARCH scalar reparameterization (cheap, per-thread)
    const float omega = softplusf(omega_raw[0]) + 1e-6f;
    const float a = sigm(alpha_raw[0]);
    const float b = sigm(beta_raw[0]) * (1.f - a * 0.99f);
    const float gam = gamma_p[0];

    const int half = lane >> 4, n = lane & 15;
#pragma unroll
    for (int ti = 0; ti < 2; ++ti)
#pragma unroll
        for (int tj = 0; tj < 2; ++tj) {
            const int gcol = colBlock + wc * 32 + tj * 16 + n;
            const float hb = Wh_b[gcol] + Uh_b[gcol];
            const float wg = Wg_w[gcol];
            const float wgb = Wg_b[gcol];
#pragma unroll
            for (int r = 0; r < 8; ++r) {
                const long long grow = rowBlock + wr * 32 + ti * 16 + half * 8 + r;
                const float ht = tanhf(acc[ti][tj][r] + hb);
                const float z  = zbuf[grow * HH + gcol];
                const float hp = h_prev[grow * HH + gcol];
                const float hhat = (1.f - z) * ht + z * hp;
                const float garch = omega + a * eps_prev[grow] + b * sig_prev[grow];
                const float gt = garch * wg + wgb;
                out_h[grow * HH + gcol] = tanhf(hhat + gam * gt);
            }
        }
}

// =====================================================================
// Kernel 3: per-row eps_sq / sigma_sq outputs
// =====================================================================
__global__ void __launch_bounds__(256)
garch_out_kernel(const float* __restrict__ x,
                 const float* __restrict__ eps_prev, const float* __restrict__ sig_prev,
                 const float* __restrict__ omega_raw, const float* __restrict__ alpha_raw,
                 const float* __restrict__ beta_raw,
                 float* __restrict__ out_eps, float* __restrict__ out_sig) {
    const int i = blockIdx.x * blockDim.x + threadIdx.x;
    if (i >= BB) return;
    const float omega = softplusf(omega_raw[0]) + 1e-6f;
    const float a = sigm(alpha_raw[0]);
    const float b = sigm(beta_raw[0]) * (1.f - a * 0.99f);
    const float x0 = x[(long long)i * INW];
    out_eps[i] = x0 * x0;
    out_sig[i] = omega + a * eps_prev[i] + b * sig_prev[i];
}

// =====================================================================
extern "C" void kernel_launch(void* const* d_in, const int* in_sizes, int n_in,
                              void* d_out, int out_size, void* d_ws, size_t ws_size,
                              hipStream_t stream) {
    (void)in_sizes; (void)n_in; (void)out_size; (void)ws_size;
    const float* x        = (const float*)d_in[0];
    const float* h_prev   = (const float*)d_in[1];
    const float* eps_prev = (const float*)d_in[2];
    const float* sig_prev = (const float*)d_in[3];
    const float* Wz_w = (const float*)d_in[4];  const float* Wz_b = (const float*)d_in[5];
    const float* Uz_w = (const float*)d_in[6];  const float* Uz_b = (const float*)d_in[7];
    const float* Wr_w = (const float*)d_in[8];  const float* Wr_b = (const float*)d_in[9];
    const float* Ur_w = (const float*)d_in[10]; const float* Ur_b = (const float*)d_in[11];
    const float* Wh_w = (const float*)d_in[12]; const float* Wh_b = (const float*)d_in[13];
    const float* Uh_w = (const float*)d_in[14]; const float* Uh_b = (const float*)d_in[15];
    const float* Wg_w = (const float*)d_in[16]; const float* Wg_b = (const float*)d_in[17];
    const float* omega_raw = (const float*)d_in[18];
    const float* alpha_raw = (const float*)d_in[19];
    const float* beta_raw  = (const float*)d_in[20];
    const float* gamma_p   = (const float*)d_in[21];

    // workspace layout (bf16 hi/lo planes + z buffer), ~118 MB total
    const long long S_A = (long long)BB * KTOT;   // elements per A plane
    const long long S_W = 3LL * HH * KTOT;        // elements per stacked W plane
    unsigned short* A1hi = (unsigned short*)d_ws;
    unsigned short* A1lo = A1hi + S_A;
    unsigned short* A3hi = A1lo + S_A;
    unsigned short* A3lo = A3hi + S_A;
    unsigned short* Whi  = A3lo + S_A;
    unsigned short* Wlo  = Whi + S_W;
    float*          zbuf = (float*)(Wlo + S_W);

    float* out_h   = (float*)d_out;
    float* out_eps = out_h + (long long)BB * HH;
    float* out_sig = out_eps + BB;

    pack_kernel<<<4096, 256, 0, stream>>>(x, h_prev, Wz_w, Uz_w, Wr_w, Ur_w, Wh_w, Uh_w,
                                          A1hi, A1lo, A3hi, A3lo, Whi, Wlo);

    dim3 grid(HH / BN, BB / BM);   // (16, 64)
    gates_kernel<<<grid, 256, 0, stream>>>(A1hi, A1lo, Whi, Wlo,
                                           Wz_b, Uz_b, Wr_b, Ur_b, h_prev,
                                           zbuf, A3hi, A3lo);
    update_kernel<<<grid, 256, 0, stream>>>(A3hi, A3lo, Whi, Wlo,
                                            Wh_b, Uh_b, Wg_w, Wg_b,
                                            h_prev, zbuf, eps_prev, sig_prev,
                                            omega_raw, alpha_raw, beta_raw, gamma_p,
                                            out_h);
    garch_out_kernel<<<(BB + 255) / 256, 256, 0, stream>>>(x, eps_prev, sig_prev,
                                                           omega_raw, alpha_raw, beta_raw,
                                                           out_eps, out_sig);
}
